// SkipEncoderModel_85169201479961
// MI455X (gfx1250) — matrix operand order repaired and checked by reading, not compiled, not run on hardware
//
#include <hip/hip_runtime.h>
#include <hip/hip_bf16.h>

// ---------------------------------------------------------------------------
// Model dims
// ---------------------------------------------------------------------------
#define BB    8
#define NS    80
#define NN    81
#define HD    256
#define HD4   1024
#define RR    (BB*NN)      // 648 rows (batch*node)
#define MSTMT (BB*NS)      // 640 rows
#define STEPS 8
#define VOCAB 30000
#define SKIPE (BB*NN*NN)   // 52488 rows of skip/keys

typedef __attribute__((ext_vector_type(16))) _Float16 v16h;
typedef __attribute__((ext_vector_type(8)))  float    v8f;

union U16h { v16h v; uint4 q[2]; };

__device__ __forceinline__ float sigmoidf_(float x) { return 1.0f / (1.0f + __expf(-x)); }

// ---------------------------------------------------------------------------
// WMMA GEMM: C[M,N] = A[M,K] (f16, row-major, lda) * Bt[N,K] (f16, row-major,
// ldb = B transposed) + bias[N]. f32 accumulate via v_wmma_f32_16x16x32_f16.
//
// Block = 4 waves (128 thr); each wave owns a 16x64 D strip (4 accumulators),
// so one A fragment feeds 4 back-to-back WMMAs per 32-deep K chunk.
// All 10 b128 loads of a chunk are issued before the first WMMA so a single
// counter wait covers the chunk. Pointers are kept as scalars indexed by k0
// so clang's addrspace inference keeps them on the GLOBAL path (an array of
// pointers demoted these to FLAT, which ties up the LDS pipe + DScnt).
// Edge rows/cols are handled by CLAMPING the source row (a clamped A row only
// pollutes the same out-of-range D row, a clamped B row only the same D col)
// and masking the stores -> branch-free hot loop.
// Fragment k order: contiguous, lane half h holds k = 16h..16h+15 in A and B (wave32).
// ---------------------------------------------------------------------------
__global__ __launch_bounds__(128) void k_wmma_gemm(
    const _Float16* __restrict__ A, int lda,
    const _Float16* __restrict__ Bt, int ldb,
    float* __restrict__ C, int ldc,
    const float* __restrict__ bias,
    int M, int N, int K)
{
    const int lane  = threadIdx.x & 31;
    const int wave  = threadIdx.x >> 5;
    const int m0    = blockIdx.x * 16;
    const int nbase = blockIdx.y * 256 + wave * 64;
    if (nbase >= N) return;

    const int half = lane >> 4;          // 0: lanes 0-15, 1: lanes 16-31
    const int lr   = lane & 15;

    int arow = m0 + lr;
    if (arow >= M) arow = M - 1;         // clamp; masked at store
    const _Float16* pa = A + (size_t)arow * lda + half * 16;  // A contiguous 16 halves

    int br0 = nbase + lr;        if (br0 >= N) br0 = N - 1;
    int br1 = nbase + 16 + lr;   if (br1 >= N) br1 = N - 1;
    int br2 = nbase + 32 + lr;   if (br2 >= N) br2 = N - 1;
    int br3 = nbase + 48 + lr;   if (br3 >= N) br3 = N - 1;
    const _Float16* pb0 = Bt + (size_t)br0 * ldb + half * 16; // B contiguous 16 halves
    const _Float16* pb1 = Bt + (size_t)br1 * ldb + half * 16;
    const _Float16* pb2 = Bt + (size_t)br2 * ldb + half * 16;
    const _Float16* pb3 = Bt + (size_t)br3 * ldb + half * 16;

    v8f acc0 = {}, acc1 = {}, acc2 = {}, acc3 = {};

    for (int k0 = 0; k0 < K; k0 += 32) {
        U16h a, b0, b1, b2, b3;
        // issue the whole chunk's loads up front (one wait covers them all)
        a.q[0]  = *(const uint4*)(pa + k0);
        a.q[1]  = *(const uint4*)(pa + k0 + 8);
        b0.q[0] = *(const uint4*)(pb0 + k0);
        b0.q[1] = *(const uint4*)(pb0 + k0 + 8);
        b1.q[0] = *(const uint4*)(pb1 + k0);
        b1.q[1] = *(const uint4*)(pb1 + k0 + 8);
        b2.q[0] = *(const uint4*)(pb2 + k0);
        b2.q[1] = *(const uint4*)(pb2 + k0 + 8);
        b3.q[0] = *(const uint4*)(pb3 + k0);
        b3.q[1] = *(const uint4*)(pb3 + k0 + 8);
        // stream B ahead (global_prefetch)
        __builtin_prefetch(pb0 + k0 + 256, 0, 1);
        __builtin_prefetch(pb1 + k0 + 256, 0, 1);
        __builtin_prefetch(pb2 + k0 + 256, 0, 1);
        __builtin_prefetch(pb3 + k0 + 256, 0, 1);

        acc0 = __builtin_amdgcn_wmma_f32_16x16x32_f16(
            false, a.v, false, b0.v, (short)0, acc0, false, false);
        acc1 = __builtin_amdgcn_wmma_f32_16x16x32_f16(
            false, a.v, false, b1.v, (short)0, acc1, false, false);
        acc2 = __builtin_amdgcn_wmma_f32_16x16x32_f16(
            false, a.v, false, b2.v, (short)0, acc2, false, false);
        acc3 = __builtin_amdgcn_wmma_f32_16x16x32_f16(
            false, a.v, false, b3.v, (short)0, acc3, false, false);
    }

    auto store_tile = [&](const v8f& acc, int nt) {
        const int n = nbase + nt * 16 + lr;
        if (n >= N) return;
        const float bv = bias ? bias[n] : 0.0f;
#pragma unroll
        for (int v = 0; v < 8; ++v) {
            const int m = m0 + v + half * 8;
            if (m < M) C[(size_t)m * ldc + n] = acc[v] + bv;
        }
    };
    store_tile(acc0, 0);
    store_tile(acc1, 1);
    store_tile(acc2, 2);
    store_tile(acc3, 3);
}

// ---------------------------------------------------------------------------
// Utility kernels
// ---------------------------------------------------------------------------
__global__ void k_zero32(float* p, int n) {
    int i = blockIdx.x * blockDim.x + threadIdx.x;
    if (i < n) p[i] = 0.0f;
}
__global__ void k_zero16(_Float16* p, int n) {
    int i = blockIdx.x * blockDim.x + threadIdx.x;
    if (i < n) p[i] = (_Float16)0.0f;
}
__global__ void k_cvt_f16(const float* __restrict__ s, _Float16* __restrict__ d, int n) {
    int i = blockIdx.x * blockDim.x + threadIdx.x;
    if (i < n) d[i] = (_Float16)s[i];
}
// dst[n*ldd + coff + k] = (f16) src[k*N + n]   (src is [K,N] row-major)
__global__ void k_transpose_cvt(const float* __restrict__ src, _Float16* __restrict__ dst,
                                int K, int N, int ldd, int coff) {
    int n = blockIdx.x * 32 + (threadIdx.x & 31);
    int k = blockIdx.y * 8 + (threadIdx.x >> 5);
    if (n < N && k < K)
        dst[(size_t)n * ldd + coff + k] = (_Float16)src[(size_t)k * N + n];
}
__global__ void k_init_p(float* p) {
    int i = blockIdx.x * blockDim.x + threadIdx.x;
    if (i < RR) p[i] = ((i % NN) == 0) ? 1.0f : 0.0f;
}

// ---------------------------------------------------------------------------
// Statement embedder cell (zero carry): h = sig(o)*tanh(sig(i)*tanh(g))
// (bias already folded into the gate GEMM)
// ---------------------------------------------------------------------------
__global__ void k_cell_nostate(const float* __restrict__ g, _Float16* __restrict__ h16, int M) {
    int idx = blockIdx.x * blockDim.x + threadIdx.x;
    if (idx >= M * HD) return;
    int r = idx >> 8, j = idx & 255;
    const float* gr = g + (size_t)r * HD4;
    float gi = gr[j], gg = gr[2 * HD + j], go = gr[3 * HD + j];
    float c = sigmoidf_(gi) * tanhf(gg);
    float h = sigmoidf_(go) * tanhf(c);
    h16[(size_t)r * HD + j] = (_Float16)h;
}

// ---------------------------------------------------------------------------
// Skip-encoder layer-0 cell: g += xWx0[b,t] (contains x@Wx0 + b0); standard
// LSTM update against post-LN carry. Writes pre-LN c0,h0 (f32) and h0 (f16)
// into cols 0..255 of the fused layer-1 A matrix.
// ---------------------------------------------------------------------------
__global__ void k_skip_cell0(const float* __restrict__ g, const float* __restrict__ xWx0,
                             int t, const float* __restrict__ c_in,
                             float* __restrict__ c_out, float* __restrict__ h_out,
                             _Float16* __restrict__ a2) {
    int idx = blockIdx.x * blockDim.x + threadIdx.x;
    if (idx >= RR * HD) return;
    int r = idx >> 8, j = idx & 255;
    int b = r / NN;
    const float* xw = xWx0 + ((size_t)b * NS + t) * HD4;
    const float* gr = g + (size_t)r * HD4;
    float gi = gr[j] + xw[j];
    float gf = gr[HD + j] + xw[HD + j];
    float gg = gr[2 * HD + j] + xw[2 * HD + j];
    float go = gr[3 * HD + j] + xw[3 * HD + j];
    float cp = c_in[(size_t)r * HD + j];
    float c = sigmoidf_(gf) * cp + sigmoidf_(gi) * tanhf(gg);
    float h = sigmoidf_(go) * tanhf(c);
    c_out[(size_t)r * HD + j] = c;
    h_out[(size_t)r * HD + j] = h;
    a2[(size_t)r * 512 + j] = (_Float16)h;
}

// ---------------------------------------------------------------------------
// Skip-encoder layer-1 cell + LayerNorm over [c0,h0,c1,h1] (1024) + keep-mask
// + skip-tile emission (pre-LN h1). One block (256 thr) per row.
// ---------------------------------------------------------------------------
__global__ __launch_bounds__(256) void k_skip_cell1_ln(
    const float* __restrict__ g,
    const float* __restrict__ c0t, const float* __restrict__ h0t,
    const float* __restrict__ c1_in,
    const float* __restrict__ ln_s, const float* __restrict__ ln_b,
    float* __restrict__ c0_out, _Float16* __restrict__ a1,
    float* __restrict__ c1_out, _Float16* __restrict__ a2,
    _Float16* __restrict__ skip16, int t)
{
    int r = blockIdx.x, j = threadIdx.x;
    const float* gr = g + (size_t)r * HD4;
    float gi = gr[j], gf = gr[HD + j], gg = gr[2 * HD + j], go = gr[3 * HD + j];
    float c1p = c1_in[(size_t)r * HD + j];
    float c1 = sigmoidf_(gf) * c1p + sigmoidf_(gi) * tanhf(gg);
    float h1 = sigmoidf_(go) * tanhf(c1);

    int b = r / NN, s = r - b * NN;
    float keep = (t >= s) ? 1.0f : 0.0f;
    // skip[b, s, t+1, :] = pre-LN h1 (masked)
    skip16[(((size_t)b * NN + s) * NN + (t + 1)) * HD + j] = (_Float16)(h1 * keep);

    float c0 = c0t[(size_t)r * HD + j];
    float h0 = h0t[(size_t)r * HD + j];

    __shared__ float red[256];
    float lsum = c0 + h0 + c1 + h1;
    float lsq = c0 * c0 + h0 * h0 + c1 * c1 + h1 * h1;
    red[j] = lsum; __syncthreads();
    for (int st = 128; st > 0; st >>= 1) { if (j < st) red[j] += red[j + st]; __syncthreads(); }
    float mu = red[0] * (1.0f / 1024.0f); __syncthreads();
    red[j] = lsq; __syncthreads();
    for (int st = 128; st > 0; st >>= 1) { if (j < st) red[j] += red[j + st]; __syncthreads(); }
    float var = red[0] * (1.0f / 1024.0f) - mu * mu;
    float inv = rsqrtf(var + 1e-6f);

    float y;
    y = (c0 - mu) * inv * ln_s[j]            + ln_b[j];
    c0_out[(size_t)r * HD + j] = y * keep;
    y = (h0 - mu) * inv * ln_s[HD + j]       + ln_b[HD + j];
    a1[(size_t)r * HD + j] = (_Float16)(y * keep);
    y = (c1 - mu) * inv * ln_s[2 * HD + j]   + ln_b[2 * HD + j];
    c1_out[(size_t)r * HD + j] = y * keep;
    y = (h1 - mu) * inv * ln_s[3 * HD + j]   + ln_b[3 * HD + j];
    a2[(size_t)r * 512 + HD + j] = (_Float16)(y * keep);
}

// ---------------------------------------------------------------------------
// logits[b,n,m] = sum_k tanh(q[b,n,k] + keys[b,n,m,k]) * w1[k] + b1, masked.
// mtype: 0 = init (m==1), 1 = mid (m>n || m==NS), 2 = final (m==NS)
// ---------------------------------------------------------------------------
__global__ void k_logits(const float* __restrict__ q, const float* __restrict__ keys,
                         const float* __restrict__ w1, const float* __restrict__ b1,
                         float* __restrict__ logits, int mtype)
{
    int idx = blockIdx.x * blockDim.x + threadIdx.x;
    if (idx >= SKIPE) return;
    int m = idx % NN;
    int bn = idx / NN;
    int n = bn % NN;
    bool valid = (mtype == 0) ? (m == 1)
               : (mtype == 2) ? (m == NS)
                              : ((m > n) || (m == NS));
    float out = -1e9f;
    if (valid) {
        const float* qr = q + (size_t)bn * HD;
        const float* kr = keys + (size_t)idx * HD;
        float acc = 0.0f;
        for (int k = 0; k < HD; ++k) acc += tanhf(qr[k] + kr[k]) * w1[k];
        out = acc + b1[0];
    }
    logits[idx] = out;
}

// softmax over m + transition mass t[b,n,m] = p[b,n] * a[m]. Block per (b,n).
__global__ __launch_bounds__(128) void k_softmax_t(const float* __restrict__ logits,
                                                   const float* __restrict__ p,
                                                   float* __restrict__ tmat)
{
    int bn = blockIdx.x, m = threadIdx.x;
    __shared__ float red[128];
    float v = (m < NN) ? logits[(size_t)bn * NN + m] : -3.0e38f;
    red[m] = v; __syncthreads();
    for (int s = 64; s > 0; s >>= 1) { if (m < s) red[m] = fmaxf(red[m], red[m + s]); __syncthreads(); }
    float mx = red[0]; __syncthreads();
    float e = (m < NN) ? __expf(v - mx) : 0.0f;
    red[m] = e; __syncthreads();
    for (int s = 64; s > 0; s >>= 1) { if (m < s) red[m] += red[m + s]; __syncthreads(); }
    float sum = red[0];
    if (m < NN) tmat[(size_t)bn * NN + m] = p[bn] * (e / sum);
}

__global__ void k_newp(const float* __restrict__ tmat, float* __restrict__ pnew,
                       float* __restrict__ denom)
{
    int idx = blockIdx.x * blockDim.x + threadIdx.x;
    if (idx >= RR) return;
    int b = idx / NN, m = idx % NN;
    float s = 0.0f;
    for (int n = 0; n < NN; ++n) s += tmat[((size_t)b * NN + n) * NN + m];
    pnew[idx] = s;
    denom[idx] = s + 1e-7f;
}

// x_in[b,m,:] = (sum_n t[b,n,m] * skip[b,n,m,:]) / denom -> ex_in0 cols 0..255
__global__ __launch_bounds__(256) void k_xin(const float* __restrict__ tmat,
                                             const _Float16* __restrict__ skip16,
                                             const float* __restrict__ denom,
                                             _Float16* __restrict__ ex_in0)
{
    int r = blockIdx.x, j = threadIdx.x;
    int b = r / NN, m = r - b * NN;
    float acc = 0.0f;
    for (int n = 0; n < NN; ++n) {
        float w = tmat[((size_t)b * NN + n) * NN + m];
        acc += w * (float)skip16[(((size_t)b * NN + n) * NN + m) * HD + j];
    }
    ex_in0[(size_t)r * 512 + j] = (_Float16)(acc / denom[r]);
}

// Propagate c/h: prop[l,b,m,:] = sum_n t[b,n,m]*state[l,b,n,:]/denom.
// kind = blockIdx.y: 0=c0 1=c1 2=h0 3=h1; h also lands (f16) in the fused A.
__global__ __launch_bounds__(256) void k_prop(const float* __restrict__ tmat,
                                              const float* __restrict__ cst,
                                              const float* __restrict__ hst,
                                              const float* __restrict__ denom,
                                              float* __restrict__ propC, float* __restrict__ propH,
                                              _Float16* __restrict__ ex_in0,
                                              _Float16* __restrict__ ex_in1)
{
    int r = blockIdx.x, kind = blockIdx.y, j = threadIdx.x;
    int b = r / NN, m = r - b * NN;
    int l = kind & 1;
    bool isH = kind >= 2;
    const float* st = (isH ? hst : cst) + (size_t)l * RR * HD;
    float acc = 0.0f;
    for (int n = 0; n < NN; ++n) {
        float w = tmat[((size_t)b * NN + n) * NN + m];
        acc += w * st[((size_t)b * NN + n) * HD + j];
    }
    acc /= denom[r];
    if (isH) {
        propH[(size_t)l * RR * HD + (size_t)r * HD + j] = acc;
        _Float16* dst = (l == 0) ? ex_in0 : ex_in1;
        dst[(size_t)r * 512 + HD + j] = (_Float16)acc;
    } else {
        propC[(size_t)l * RR * HD + (size_t)r * HD + j] = acc;
    }
}

// Execution LSTM cell; also refreshes the f16 hcat=[c0,h0,c1,h1] for next q.
__global__ void k_ex_cell(const float* __restrict__ g, const float* __restrict__ cprop,
                          float* __restrict__ c_out, float* __restrict__ h_out,
                          _Float16* __restrict__ hcat16, int coff, int hoff,
                          _Float16* __restrict__ next_in)
{
    int idx = blockIdx.x * blockDim.x + threadIdx.x;
    if (idx >= RR * HD) return;
    int r = idx >> 8, j = idx & 255;
    const float* gr = g + (size_t)r * HD4;
    float gi = gr[j], gf = gr[HD + j], gg = gr[2 * HD + j], go = gr[3 * HD + j];
    float c = sigmoidf_(gf) * cprop[(size_t)r * HD + j] + sigmoidf_(gi) * tanhf(gg);
    float h = sigmoidf_(go) * tanhf(c);
    c_out[(size_t)r * HD + j] = c;
    h_out[(size_t)r * HD + j] = h;
    hcat16[(size_t)r * HD4 + coff + j] = (_Float16)c;
    hcat16[(size_t)r * HD4 + hoff + j] = (_Float16)h;
    if (next_in) next_in[(size_t)r * 512 + j] = (_Float16)h;
}

__global__ void k_gather_exit(const float* __restrict__ hL, const int* __restrict__ exit_idx,
                              _Float16* __restrict__ out16)
{
    int b = blockIdx.x, j = threadIdx.x;
    int e = exit_idx[b];
    out16[(size_t)b * HD + j] = (_Float16)hL[((size_t)b * NN + e) * HD + j];
}

// ---------------------------------------------------------------------------
// Host side
// ---------------------------------------------------------------------------
static inline void gemm(hipStream_t s, const _Float16* A, int lda, const _Float16* Bt, int ldb,
                        float* C, int ldc, const float* bias, int M, int N, int K)
{
    dim3 g((M + 15) / 16, (N + 255) / 256);
    k_wmma_gemm<<<g, 128, 0, s>>>(A, lda, Bt, ldb, C, ldc, bias, M, N, K);
}

extern "C" void kernel_launch(void* const* d_in, const int* in_sizes, int n_in,
                              void* d_out, int out_size, void* d_ws, size_t ws_size,
                              hipStream_t stream)
{
    (void)in_sizes; (void)n_in; (void)out_size; (void)ws_size;

    const float* node_emb = (const float*)d_in[0];
    const int*   exit_idx = (const int*)d_in[10];
    const float* st_Wx = (const float*)d_in[12];
    const float* st_b  = (const float*)d_in[14];
    const float* sk_Wx = (const float*)d_in[15];
    const float* sk_Wh = (const float*)d_in[16];
    const float* sk_b  = (const float*)d_in[17];
    const float* ln_s  = (const float*)d_in[18];
    const float* ln_b  = (const float*)d_in[19];
    const float* ex_Wx = (const float*)d_in[20];
    const float* ex_Wh = (const float*)d_in[21];
    const float* ex_b  = (const float*)d_in[22];
    const float* wk    = (const float*)d_in[23];
    const float* bk    = (const float*)d_in[24];
    const float* wsm   = (const float*)d_in[25];
    const float* bs    = (const float*)d_in[26];
    const float* w1    = (const float*)d_in[27];
    const float* b1    = (const float*)d_in[28];
    const float* wo    = (const float*)d_in[29];
    const float* bo    = (const float*)d_in[30];
    float* out = (float*)d_out;

    // ---- workspace carve-up ----
    char* base = (char*)d_ws;
    size_t off = 0;
    auto alloc = [&](size_t bytes) -> char* {
        off = (off + 255) & ~(size_t)255;
        char* p = base + off;
        off += bytes;
        return p;
    };
    const size_t WH = HD * HD4;  // 256*1024 per weight layer

    _Float16* stWx0T   = (_Float16*)alloc(WH * 2);
    _Float16* stWx1T   = (_Float16*)alloc(WH * 2);
    _Float16* skWx0T   = (_Float16*)alloc(WH * 2);
    _Float16* skWh0T   = (_Float16*)alloc(WH * 2);
    _Float16* skWcat1T = (_Float16*)alloc(HD4 * 512 * 2);
    _Float16* exW0T    = (_Float16*)alloc(HD4 * 512 * 2);
    _Float16* exW1T    = (_Float16*)alloc(HD4 * 512 * 2);
    _Float16* wkT      = (_Float16*)alloc(HD * HD4 * 2);
    _Float16* wsT      = (_Float16*)alloc(HD * HD * 2);
    _Float16* woT      = (_Float16*)alloc((size_t)VOCAB * HD * 2);

    _Float16* node16  = (_Float16*)alloc((size_t)MSTMT * HD * 2);
    _Float16* bufh16  = (_Float16*)alloc((size_t)MSTMT * HD * 2);
    _Float16* stmt16  = (_Float16*)alloc((size_t)MSTMT * HD * 2);
    float*    gbuf    = (float*)alloc((size_t)RR * HD4 * 4);
    float*    xWx0    = (float*)alloc((size_t)MSTMT * HD4 * 4);

    float*    stC0    = (float*)alloc((size_t)RR * HD * 4);
    float*    stC1    = (float*)alloc((size_t)RR * HD * 4);
    float*    tmpC0   = (float*)alloc((size_t)RR * HD * 4);
    float*    tmpH0   = (float*)alloc((size_t)RR * HD * 4);
    _Float16* skA1    = (_Float16*)alloc((size_t)RR * HD * 2);
    _Float16* skA2    = (_Float16*)alloc((size_t)RR * 512 * 2);

    _Float16* skip16  = (_Float16*)alloc((size_t)SKIPE * HD * 2);
    float*    keys    = (float*)alloc((size_t)SKIPE * HD * 4);

    float*    qbuf    = (float*)alloc((size_t)RR * HD * 4);
    float*    logits  = (float*)alloc((size_t)SKIPE * 4);
    float*    tmat    = (float*)alloc((size_t)SKIPE * 4);
    float*    pA      = (float*)alloc((size_t)RR * 4);
    float*    pB      = (float*)alloc((size_t)RR * 4);
    float*    denom   = (float*)alloc((size_t)RR * 4);

    float*    exC     = (float*)alloc((size_t)2 * RR * HD * 4);
    float*    exH     = (float*)alloc((size_t)2 * RR * HD * 4);
    float*    propC   = (float*)alloc((size_t)2 * RR * HD * 4);
    float*    propH   = (float*)alloc((size_t)2 * RR * HD * 4);
    _Float16* hcat16  = (_Float16*)alloc((size_t)RR * HD4 * 2);
    _Float16* ex_in0  = (_Float16*)alloc((size_t)RR * 512 * 2);
    _Float16* ex_in1  = (_Float16*)alloc((size_t)RR * 512 * 2);
    _Float16* hexit16 = (_Float16*)alloc((size_t)BB * HD * 2);

    auto tcvt = [&](const float* src, _Float16* dst, int K, int N, int ldd, int coff) {
        dim3 g((N + 31) / 32, (K + 7) / 8);
        k_transpose_cvt<<<g, 256, 0, stream>>>(src, dst, K, N, ldd, coff);
    };
    auto zero32 = [&](float* p, int n) {
        k_zero32<<<(n + 255) / 256, 256, 0, stream>>>(p, n);
    };
    auto zero16 = [&](_Float16* p, int n) {
        k_zero16<<<(n + 255) / 256, 256, 0, stream>>>(p, n);
    };

    // ---- weight prep (f32 [K,N] -> f16 [N,K]) ----
    tcvt(st_Wx,            stWx0T, HD, HD4, HD, 0);
    tcvt(st_Wx + WH,       stWx1T, HD, HD4, HD, 0);
    tcvt(sk_Wx,            skWx0T, HD, HD4, HD, 0);
    tcvt(sk_Wh,            skWh0T, HD, HD4, HD, 0);
    tcvt(sk_Wx + WH,       skWcat1T, HD, HD4, 512, 0);    // [Wx1 ; Wh1] fused K=512
    tcvt(sk_Wh + WH,       skWcat1T, HD, HD4, 512, HD);
    tcvt(ex_Wx,            exW0T, HD, HD4, 512, 0);
    tcvt(ex_Wh,            exW0T, HD, HD4, 512, HD);
    tcvt(ex_Wx + WH,       exW1T, HD, HD4, 512, 0);
    tcvt(ex_Wh + WH,       exW1T, HD, HD4, 512, HD);
    tcvt(wk,               wkT, HD4, HD, HD4, 0);
    tcvt(wsm,              wsT, HD, HD, HD, 0);
    tcvt(wo,               woT, HD, VOCAB, HD, 0);

    // ---- deterministic state init ----
    zero32(stC0, RR * HD); zero32(stC1, RR * HD);
    zero16(skA1, RR * HD); zero16(skA2, RR * 512);
    zero16(skip16, SKIPE * HD);
    zero16(hcat16, RR * HD4);
    zero32(exC, 2 * RR * HD); zero32(exH, 2 * RR * HD);
    k_init_p<<<(RR + 255) / 256, 256, 0, stream>>>(pA);

    // ---- statement embedder (single step, zero carry: Wh drops out) ----
    k_cvt_f16<<<(MSTMT * HD + 255) / 256, 256, 0, stream>>>(node_emb, node16, MSTMT * HD);
    gemm(stream, node16, HD, stWx0T, HD, gbuf, HD4, st_b, MSTMT, HD4, HD);
    k_cell_nostate<<<(MSTMT * HD + 255) / 256, 256, 0, stream>>>(gbuf, bufh16, MSTMT);
    gemm(stream, bufh16, HD, stWx1T, HD, gbuf, HD4, st_b + HD4, MSTMT, HD4, HD);
    k_cell_nostate<<<(MSTMT * HD + 255) / 256, 256, 0, stream>>>(gbuf, stmt16, MSTMT);

    // hoisted layer-0 input projection for the skip scan: xWx0 = stmt@Wx0 + b0
    gemm(stream, stmt16, HD, skWx0T, HD, xWx0, HD4, sk_b, MSTMT, HD4, HD);

    // ---- skip encoder scan: 80 sequential steps over 648 rows ----
    for (int t = 0; t < NS; ++t) {
        gemm(stream, skA1, HD, skWh0T, HD, gbuf, HD4, nullptr, RR, HD4, HD);
        k_skip_cell0<<<(RR * HD + 255) / 256, 256, 0, stream>>>(
            gbuf, xWx0, t, stC0, tmpC0, tmpH0, skA2);
        gemm(stream, skA2, 512, skWcat1T, 512, gbuf, HD4, sk_b + HD4, RR, HD4, 512);
        k_skip_cell1_ln<<<RR, 256, 0, stream>>>(
            gbuf, tmpC0, tmpH0, stC1, ln_s, ln_b, stC0, skA1, stC1, skA2, skip16, t);
    }

    // ---- keys = skip @ ws + bs ----
    gemm(stream, skip16, HD, wsT, HD, keys, HD, bs, SKIPE, HD, HD);

    // ---- execution loop ----
    float* pcur = pA;
    float* pnxt = pB;
    for (int step = 0; step < STEPS; ++step) {
        const int mtype = (step == 0) ? 0 : (step == STEPS - 1) ? 2 : 1;
        gemm(stream, hcat16, HD4, wkT, HD4, qbuf, HD, bk, RR, HD, HD4);
        k_logits<<<(SKIPE + 255) / 256, 256, 0, stream>>>(qbuf, keys, w1, b1, logits, mtype);
        k_softmax_t<<<RR, 128, 0, stream>>>(logits, pcur, tmat);
        k_newp<<<(RR + 255) / 256, 256, 0, stream>>>(tmat, pnxt, denom);
        k_xin<<<RR, 256, 0, stream>>>(tmat, skip16, denom, ex_in0);
        k_prop<<<dim3(RR, 4), 256, 0, stream>>>(tmat, exC, exH, denom, propC, propH,
                                                ex_in0, ex_in1);
        gemm(stream, ex_in0, 512, exW0T, 512, gbuf, HD4, ex_b, RR, HD4, 512);
        k_ex_cell<<<(RR * HD + 255) / 256, 256, 0, stream>>>(
            gbuf, propC, exC, exH, hcat16, 0, HD, ex_in1);
        gemm(stream, ex_in1, 512, exW1T, 512, gbuf, HD4, ex_b + HD4, RR, HD4, 512);
        k_ex_cell<<<(RR * HD + 255) / 256, 256, 0, stream>>>(
            gbuf, propC + (size_t)RR * HD, exC + (size_t)RR * HD, exH + (size_t)RR * HD,
            hcat16, 2 * HD, 3 * HD, nullptr);
        float* tmp = pcur; pcur = pnxt; pnxt = tmp;
    }

    // ---- output head ----
    k_gather_exit<<<BB, 256, 0, stream>>>(exH + (size_t)RR * HD, exit_idx, hexit16);
    gemm(stream, hexit16, HD, woT, HD, out, VOCAB, bo, BB, VOCAB, HD);
}
